// MultiHeadAttention_48747878809780
// MI455X (gfx1250) — matrix operand order repaired and checked by reading, not compiled, not run on hardware
//
#include <hip/hip_runtime.h>
#include <hip/hip_bf16.h>
#include <math.h>

// Problem constants (from reference)
#define S_LEN 4096
#define HID   1024
#define NHEAD 16
#define DH    64
#define QKV_E (3 * HID)   // 3072

typedef __bf16 bf16_t;
typedef __attribute__((ext_vector_type(16))) __bf16 v16bf;
typedef __attribute__((ext_vector_type(8)))  __bf16 v8bf;
typedef __attribute__((ext_vector_type(8)))  float  v8f;
typedef int v4i_t __attribute__((vector_size(4 * sizeof(int))));  // b128 payload

// CDNA5 async global->LDS path (ASYNCcnt), if the toolchain exposes it.
#if defined(__has_builtin)
#if __has_builtin(__builtin_amdgcn_global_load_async_to_lds_b128) && \
    __has_builtin(__builtin_amdgcn_s_wait_asynccnt)
#define HAVE_ASYNC_LDS 1
#endif
#endif
#ifndef HAVE_ASYNC_LDS
#define HAVE_ASYNC_LDS 0
#endif

union FragU { v16bf v; v8bf h[2]; };

__device__ __forceinline__ int lane_id() { return (int)(threadIdx.x & 31u); }

// ---------------------------------------------------------------------------
// WMMA fragment loaders, CDNA5 ISA §7.12.2 VGPR layouts (wave32).
// A (16x32 bf16): lane L<16: M=L, K groups {0..7},{16..23};
//                 lane L>=16: M=L-16, K groups {8..15},{24..31}.
// ---------------------------------------------------------------------------
__device__ __forceinline__ v16bf load_frag_a(const bf16_t* p, int ld, int m0, int k0) {
  const int lane = lane_id();
  const int m    = m0 + (lane & 15);
  const int hh   = lane >> 4;
  const bf16_t* r = p + (size_t)m * (size_t)ld + (size_t)(k0 + hh * 8);
  FragU u;
  u.h[0] = *reinterpret_cast<const v8bf*>(r);
  u.h[1] = *reinterpret_cast<const v8bf*>(r + 16);
  return u.v;
}

// B (32x16 bf16): lane L<16: N=L, K groups {0..7},{16..23}; lane L>=16: N=L-16, K groups {8..15},{24..31}.
// Element B(k,n) = p[(n0+n)*ld + k0+k]  ("torch Linear" weight layout)
__device__ __forceinline__ v16bf load_frag_b(const bf16_t* p, int ld, int n0, int k0) {
  const int lane = lane_id();
  const int n    = n0 + (lane & 15);
  const int hh   = lane >> 4;
  const bf16_t* r = p + (size_t)n * (size_t)ld + (size_t)(k0 + hh * 8);
  FragU u;
  u.h[0] = *reinterpret_cast<const v8bf*>(r);
  u.h[1] = *reinterpret_cast<const v8bf*>(r + 16);
  return u.v;
}

__device__ __forceinline__ v8f wmma_bf16(v16bf a, v16bf b, v8f c) {
  return __builtin_amdgcn_wmma_f32_16x16x32_bf16(
      /*neg_a=*/false, a, /*neg_b=*/false, b,
      /*c_mod=*/(short)0, c, /*reuse_a=*/false, /*reuse_b=*/false);
}

__device__ __forceinline__ void wait_async_done() {
#if HAVE_ASYNC_LDS
  __builtin_amdgcn_s_wait_asynccnt(0);
#endif
}

// One 16-byte global->LDS transfer (async when available)
__device__ __forceinline__ void copy16(const bf16_t* gp, bf16_t* lp) {
#if HAVE_ASYNC_LDS
  __builtin_amdgcn_global_load_async_to_lds_b128((v4i_t*)gp, (v4i_t*)lp, 0, 0);
#else
  *reinterpret_cast<v8bf*>(lp) = *reinterpret_cast<const v8bf*>(gp);
#endif
}

// ---------------------------------------------------------------------------
// fp32 -> bf16 conversion
// ---------------------------------------------------------------------------
__global__ void f32_to_bf16_kernel(const float* __restrict__ in,
                                   bf16_t* __restrict__ out, int n) {
  int i = blockIdx.x * blockDim.x + threadIdx.x;
  if (i < n) out[i] = (bf16_t)in[i];
}

// ---------------------------------------------------------------------------
// Stage a 64x32 bf16 tile (rows r0..r0+63 of src with leading dim ld,
// cols k..k+31) into LDS [64][32]. 256 x 16B transfers, 128 threads x 2.
// ---------------------------------------------------------------------------
__device__ __forceinline__ void stage_tile_64x32(const bf16_t* __restrict__ src,
                                                 int ld, int r0, int k,
                                                 bf16_t* lds) {
  const int t = (int)threadIdx.x;  // 0..127
#pragma unroll
  for (int i = 0; i < 2; ++i) {
    const int tr  = t + i * 128;     // 0..255
    const int row = tr >> 2;         // 0..63
    const int seg = (tr & 3) * 8;    // 8 bf16 = 16 B
    copy16(src + (size_t)(r0 + row) * (size_t)ld + (size_t)(k + seg),
           lds + row * 32 + seg);
  }
}

// Stage a 64x64 bf16 tile (rows r0..r0+63, cols c0..c0+63) into LDS [64][64].
// 512 x 16B transfers, 128 threads x 4.
__device__ __forceinline__ void stage_tile_64x64(const bf16_t* __restrict__ src,
                                                 int ld, int r0, int c0,
                                                 bf16_t* lds) {
  const int t = (int)threadIdx.x;  // 0..127
#pragma unroll
  for (int i = 0; i < 4; ++i) {
    const int tr  = t + i * 128;     // 0..511
    const int row = tr >> 3;         // 0..63
    const int seg = (tr & 7) * 8;    // 8 bf16 = 16 B
    copy16(src + (size_t)(r0 + row) * (size_t)ld + (size_t)(c0 + seg),
           lds + row * 64 + seg);
  }
}

// ---------------------------------------------------------------------------
// C[m,n] = sum_k A[m,k] * B[n,k]   (A: MxK, B: NxK row-major, C f32)
// Block = 128 threads = 4 waves; block tile 128x64; wave tile 32x64.
// B double-buffered in LDS via async-to-LDS; A software-pipelined in registers.
// ---------------------------------------------------------------------------
__global__ __launch_bounds__(128)
void gemm_bf16_kernel(const bf16_t* __restrict__ A, const bf16_t* __restrict__ B,
                      float* __restrict__ C, int M, int N, int K) {
  __shared__ __align__(32) bf16_t ldsB[2][64][32];   // 8 KB double buffer

  const int wave = threadIdx.x >> 5;
  const int m0   = blockIdx.y * 128 + wave * 32;
  const int n0   = blockIdx.x * 64;
  const int lane = lane_id();
  const int half = lane >> 4;
  const int nn   = lane & 15;

  v8f acc[2][4] = {};

  stage_tile_64x32(B, K, n0, 0, &ldsB[0][0][0]);
  v16bf a_cur[2];
  a_cur[0] = load_frag_a(A, K, m0,      0);
  a_cur[1] = load_frag_a(A, K, m0 + 16, 0);
  wait_async_done();
  __syncthreads();

  for (int k = 0; k < K; k += 32) {
    const int  cur  = (k >> 5) & 1;
    const bool more = (k + 32) < K;

    if (more) stage_tile_64x32(B, K, n0, k + 32, &ldsB[cur ^ 1][0][0]);
    v16bf a_nxt0, a_nxt1;
    if (more) {
      a_nxt0 = load_frag_a(A, K, m0,      k + 32);
      a_nxt1 = load_frag_a(A, K, m0 + 16, k + 32);
    }

    // preload all B fragments, then issue the 8 WMMAs back-to-back
    v16bf b[4];
#pragma unroll
    for (int j = 0; j < 4; ++j)
      b[j] = load_frag_b(&ldsB[cur][0][0], 32, j * 16, 0);
#pragma unroll
    for (int j = 0; j < 4; ++j)
#pragma unroll
      for (int i = 0; i < 2; ++i)
        acc[i][j] = wmma_bf16(i ? a_cur[1] : a_cur[0], b[j], acc[i][j]);

    if (more) {
      wait_async_done();
      a_cur[0] = a_nxt0;
      a_cur[1] = a_nxt1;
    }
    __syncthreads();
  }

#pragma unroll
  for (int i = 0; i < 2; ++i)
#pragma unroll
    for (int j = 0; j < 4; ++j)
#pragma unroll
      for (int g = 0; g < 8; ++g)
        C[(size_t)(m0 + i * 16 + g + 8 * half) * N + (n0 + j * 16 + nn)] =
            acc[i][j][g];
}

// ---------------------------------------------------------------------------
// RoPE + pack: read qkv fp32 [S, 3072]; emit
//   Q  [H][S][64] bf16 (RoPE'd, pre-scaled by 1/sqrt(64))
//   Kp [H][S][64] bf16 (RoPE'd)
//   VT [H][64][S] bf16 (transposed so P@V B-fragments are contiguous)
// ---------------------------------------------------------------------------
__global__ void rope_pack_kernel(const float* __restrict__ qkv,
                                 bf16_t* __restrict__ Q, bf16_t* __restrict__ Kp,
                                 bf16_t* __restrict__ VT) {
  const int s = blockIdx.x;
  const int h = blockIdx.y;
  const int d = threadIdx.x;           // 0..63
  const float* row = qkv + (size_t)s * QKV_E;
  const int base = h * DH;

  const float q = row[base + d];
  const float k = row[HID + base + d];
  const float v = row[2 * HID + base + d];

  const int   dr  = d & 31;
  const float inv = __expf(-(float)dr * 0.28782313662425572f);  // 10000^{-dr/32}
  const float ang = (float)s * inv;
  const float c  = __cosf(ang);
  const float sn = __sinf(ang);
  const float qrot = (d < 32) ? -row[base + d + 32]       : row[base + d - 32];
  const float krot = (d < 32) ? -row[HID + base + d + 32] : row[HID + base + d - 32];

  const float qr = q * c + qrot * sn;
  const float kr = k * c + krot * sn;

  Q [((size_t)h * S_LEN + s) * DH + d] = (bf16_t)(qr * 0.125f);  // 1/sqrt(64)
  Kp[((size_t)h * S_LEN + s) * DH + d] = (bf16_t)kr;
  VT[((size_t)h * DH + d) * S_LEN + s] = (bf16_t)v;
}

// ---------------------------------------------------------------------------
// FlashAttention (causal), one (64-row q-tile, head) per block.
// Block = 128 threads = 4 waves; wave w owns q rows [qt*64 + 16w, +16).
// K and V tiles staged once per block in double-buffered LDS (async DMA
// overlapped with the current tile's WMMAs + softmax); P tile bounced via LDS.
// ---------------------------------------------------------------------------
__global__ __launch_bounds__(128)
void flash_attn_kernel(const bf16_t* __restrict__ Q, const bf16_t* __restrict__ Kp,
                       const bf16_t* __restrict__ VT, bf16_t* __restrict__ Oa) {
  __shared__ __align__(32) bf16_t ldsK[2][64][DH];   // 16 KB
  __shared__ __align__(32) bf16_t ldsV[2][64][DH];   // 16 KB  (rows = d, cols = key)
  __shared__ __align__(32) bf16_t ldsP[4][16][DH];   //  8 KB

  const int qt   = blockIdx.x;
  const int h    = blockIdx.y;
  const int wave = threadIdx.x >> 5;
  const int lane = lane_id();
  const int half = lane >> 4;
  const int nn   = lane & 15;
  const int mq   = qt * 64 + wave * 16;

  const bf16_t* Qh = Q  + (size_t)h * S_LEN * DH;
  const bf16_t* Kh = Kp + (size_t)h * S_LEN * DH;
  const bf16_t* Vh = VT + (size_t)h * DH * S_LEN;   // [64][S]

  v16bf qfrag[2];
  qfrag[0] = load_frag_a(Qh, DH, mq, 0);
  qfrag[1] = load_frag_a(Qh, DH, mq, 32);

  v8f o[4] = {};
  float rmax[8], rsum[8];
#pragma unroll
  for (int g = 0; g < 8; ++g) { rmax[g] = -1e30f; rsum[g] = 0.0f; }

  // prologue: stage first K/V tile
  stage_tile_64x64(Kh, DH, 0, 0, &ldsK[0][0][0]);   // K rows 0..63, all 64 d
  stage_tile_64x64(Vh, S_LEN, 0, 0, &ldsV[0][0][0]); // V^T rows d=0..63, keys 0..63
  wait_async_done();
  __syncthreads();

  for (int kt = 0; kt <= qt; ++kt) {
    const int  n0   = kt * 64;
    const int  cur  = kt & 1;
    const bool more = kt < qt;

    // overlap next K/V tile DMA with this tile's compute
    if (more) {
      stage_tile_64x64(Kh, DH, n0 + 64, 0, &ldsK[cur ^ 1][0][0]);
      stage_tile_64x64(Vh, S_LEN, 0, n0 + 64, &ldsV[cur ^ 1][0][0]);
    }

    // ---- S = Q K^T (pre-scaled) : 16x64 per wave, K frags from LDS -------
    v8f sc[4] = {};
#pragma unroll
    for (int ks = 0; ks < 2; ++ks) {
      v16bf b[4];
#pragma unroll
      for (int j = 0; j < 4; ++j)
        b[j] = load_frag_b(&ldsK[cur][0][0], DH, j * 16, ks * 32);
#pragma unroll
      for (int j = 0; j < 4; ++j)
        sc[j] = wmma_bf16(qfrag[ks], b[j], sc[j]);
    }

    // ---- causal mask on the diagonal tile --------------------------------
    if (kt == qt) {
#pragma unroll
      for (int j = 0; j < 4; ++j)
#pragma unroll
        for (int g = 0; g < 8; ++g)
          if (n0 + j * 16 + nn > mq + g + 8 * half) sc[j][g] = -1e30f;
    }

    // ---- online softmax ---------------------------------------------------
#pragma unroll
    for (int g = 0; g < 8; ++g) {
      float mx = sc[0][g];
#pragma unroll
      for (int j = 1; j < 4; ++j) mx = fmaxf(mx, sc[j][g]);
#pragma unroll
      for (int off = 1; off < 16; off <<= 1) mx = fmaxf(mx, __shfl_xor(mx, off, 32));
      const float nm   = fmaxf(rmax[g], mx);
      const float corr = __expf(rmax[g] - nm);
      rmax[g] = nm;
      float part = 0.0f;
#pragma unroll
      for (int j = 0; j < 4; ++j) {
        const float pv = __expf(sc[j][g] - nm);
        sc[j][g] = pv;
        part += pv;
      }
#pragma unroll
      for (int off = 1; off < 16; off <<= 1) part += __shfl_xor(part, off, 32);
      rsum[g] = rsum[g] * corr + part;
#pragma unroll
      for (int j = 0; j < 4; ++j) o[j][g] *= corr;
    }

    // ---- P -> LDS (bf16), own-wave strip only ----------------------------
#pragma unroll
    for (int j = 0; j < 4; ++j)
#pragma unroll
      for (int g = 0; g < 8; ++g)
        ldsP[wave][g + 8 * half][j * 16 + nn] = (bf16_t)sc[j][g];

    // ---- O += P V  (P from own LDS strip, V frags from staged LDS) -------
#pragma unroll
    for (int ks = 0; ks < 2; ++ks) {
      v16bf pa = load_frag_a(&ldsP[wave][0][0], DH, 0, ks * 32);
      v16bf vb[4];
#pragma unroll
      for (int j = 0; j < 4; ++j)
        vb[j] = load_frag_b(&ldsV[cur][0][0], DH, j * 16, ks * 32);
#pragma unroll
      for (int j = 0; j < 4; ++j)
        o[j] = wmma_bf16(pa, vb[j], o[j]);
    }

    if (more) wait_async_done();   // own transfers for nxt buffers complete
    __syncthreads();               // all waves staged nxt & done reading cur
  }

  // ---- normalize + store attn output [S][HID] bf16 -----------------------
#pragma unroll
  for (int g = 0; g < 8; ++g) {
    const float inv_l = 1.0f / rsum[g];
    const int m = mq + g + 8 * half;
#pragma unroll
    for (int j = 0; j < 4; ++j)
      Oa[(size_t)m * HID + h * DH + j * 16 + nn] = (bf16_t)(o[j][g] * inv_l);
  }
}

// ---------------------------------------------------------------------------
// Host-side orchestration
// ---------------------------------------------------------------------------
extern "C" void kernel_launch(void* const* d_in, const int* in_sizes, int n_in,
                              void* d_out, int out_size, void* d_ws, size_t ws_size,
                              hipStream_t stream) {
  (void)in_sizes; (void)n_in; (void)out_size; (void)ws_size;
  const float* x     = (const float*)d_in[0];   // [1, 4096, 1024]
  const float* w_qkv = (const float*)d_in[1];   // [3072, 1024]
  const float* w_out = (const float*)d_in[2];   // [1024, 1024]
  float*       out   = (float*)d_out;           // [1, 4096, 1024]

  char* ws = (char*)d_ws;
  bf16_t* xb   = (bf16_t*)(ws);                         //  8 MiB  x bf16
  bf16_t* wqb  = (bf16_t*)(ws + (size_t)8  * (1u<<20)); //  6 MiB  w_qkv bf16
  bf16_t* wob  = (bf16_t*)(ws + (size_t)14 * (1u<<20)); //  2 MiB  w_out bf16
  float*  qkvf = (float*) (ws + (size_t)16 * (1u<<20)); // 48 MiB  qkv f32
  bf16_t* Qb   = (bf16_t*)(ws + (size_t)64 * (1u<<20)); //  8 MiB
  bf16_t* Kb   = (bf16_t*)(ws + (size_t)72 * (1u<<20)); //  8 MiB
  bf16_t* VTb  = (bf16_t*)(ws + (size_t)80 * (1u<<20)); //  8 MiB
  bf16_t* Ab   = (bf16_t*)(ws + (size_t)16 * (1u<<20)); // reuses qkvf region

  {
    int n;
    n = S_LEN * HID;
    f32_to_bf16_kernel<<<(n + 255) / 256, 256, 0, stream>>>(x, xb, n);
    n = QKV_E * HID;
    f32_to_bf16_kernel<<<(n + 255) / 256, 256, 0, stream>>>(w_qkv, wqb, n);
    n = HID * HID;
    f32_to_bf16_kernel<<<(n + 255) / 256, 256, 0, stream>>>(w_out, wob, n);
  }

  // qkv = x @ w_qkv^T : [4096, 3072] fp32
  gemm_bf16_kernel<<<dim3(QKV_E / 64, S_LEN / 128), 128, 0, stream>>>(
      xb, wqb, qkvf, S_LEN, QKV_E, HID);

  // RoPE + pack Q/K/V^T (bf16, head-major)
  rope_pack_kernel<<<dim3(S_LEN, NHEAD), DH, 0, stream>>>(qkvf, Qb, Kb, VTb);

  // causal FlashAttention -> attn bf16 [4096, 1024]
  flash_attn_kernel<<<dim3(S_LEN / 64, NHEAD), 128, 0, stream>>>(Qb, Kb, VTb, Ab);

  // out = attn @ w_out^T : fp32 to d_out
  gemm_bf16_kernel<<<dim3(HID / 64, S_LEN / 128), 128, 0, stream>>>(
      Ab, wob, out, S_LEN, HID, HID);
}
